// QuaternionSpikeAttention_59906203844672
// MI455X (gfx1250) — compile-verified, matrix operand order checked by reading, not run on hardware
//
#include <hip/hip_runtime.h>
#include <hip/hip_bf16.h>
#include <math.h>

#define D_MODEL 1024
#define N_HEADS 16
#define HEAD_DIM 64
#define BATCH 2
#define SEQ 2048
#define MTOT (BATCH * SEQ) /* 4096 */

typedef __attribute__((ext_vector_type(16))) _Float16 v16h;
typedef __attribute__((ext_vector_type(8)))  _Float16 v8h;
typedef __attribute__((ext_vector_type(8)))  float    v8f;

// ---------------------------------------------------------------------------
// Fragment load: A/B fragment for v_wmma_f32_16x16x32_f16.
// Lane L (row/col = L%16, half = L/16) holds K pairs at
//   g*16 + half*8 + {0..7}  for g = 0,1  -> two contiguous 16B chunks.
// ---------------------------------------------------------------------------
__device__ __forceinline__ v16h frag_ld(const _Float16* p0, const _Float16* p1) {
  v16h f;
  *(v8h*)&f       = *(const v8h*)p0;
  *((v8h*)&f + 1) = *(const v8h*)p1;
  return f;
}

__device__ __forceinline__ v8f wmma16(v16h a, v16h b, v8f c) {
  return __builtin_amdgcn_wmma_f32_16x16x32_f16(
      /*neg_a=*/false, a, /*neg_b=*/false, b,
      /*c_mod=*/(short)0, c, /*reuse_a=*/false, /*reuse_b=*/false);
}

// ---------------------------------------------------------------------------
// Prep: f32 -> f16 cast
// ---------------------------------------------------------------------------
__global__ void cvt_h(const float* __restrict__ in, _Float16* __restrict__ out, int n) {
  int i = blockIdx.x * blockDim.x + threadIdx.x;
  if (i < n) out[i] = (_Float16)in[i];
}

// ---------------------------------------------------------------------------
// Prep: build 1024x1024 Hamilton-product weight matrix in f16.
// Row blocks: [r,-i,-j,-k; i,r,-k,j; j,k,r,-i; k,-j,i,r]
// ---------------------------------------------------------------------------
__global__ void build_qweight(const float* __restrict__ wr, const float* __restrict__ wi,
                              const float* __restrict__ wj, const float* __restrict__ wk,
                              _Float16* __restrict__ W) {
  int idx = blockIdx.x * blockDim.x + threadIdx.x;
  if (idx >= D_MODEL * D_MODEL) return;
  int row = idx >> 10, col = idx & 1023;
  int br = row >> 8, bc = col >> 8;
  int ri = row & 255, ci = col & 255;
  const int   ctab[4][4] = {{0,1,2,3},{1,0,3,2},{2,3,0,1},{3,2,1,0}};
  const float stab[4][4] = {{1.f,-1.f,-1.f,-1.f},{1.f,1.f,-1.f,1.f},
                            {1.f,1.f,1.f,-1.f},{1.f,-1.f,1.f,1.f}};
  int c = ctab[br][bc];
  const float* src = (c == 0) ? wr : (c == 1) ? wi : (c == 2) ? wj : wk;
  W[idx] = (_Float16)(stab[br][bc] * src[ri * 256 + ci]);
}

// ---------------------------------------------------------------------------
// Density-modulated threshold: thr[b,n] = softplus(t) + 0.1 * MLP(density)
// ---------------------------------------------------------------------------
__global__ void density_thr(const float* __restrict__ df, const float* __restrict__ w1,
                            const float* __restrict__ b1, const float* __restrict__ w2,
                            const float* __restrict__ b2, const float* __restrict__ t,
                            float* __restrict__ thr, int n) {
  int i = blockIdx.x * blockDim.x + threadIdx.x;
  if (i >= n) return;
  float x0 = df[i * 3 + 0], x1 = df[i * 3 + 1], x2 = df[i * 3 + 2];
  float acc = b2[0];
#pragma unroll
  for (int j = 0; j < 16; ++j) {
    float h = w1[j * 3 + 0] * x0 + w1[j * 3 + 1] * x1 + w1[j * 3 + 2] * x2 + b1[j];
    float g = 0.5f * h * (1.f + erff(h * 0.70710678118654752f)); // exact gelu
    acc += w2[j] * g;
  }
  float base = log1pf(expf(t[0])); // softplus
  thr[i] = base + acc * 0.1f;
}

// ---------------------------------------------------------------------------
// WMMA GEMM: C[m,n] = sum_k A[m,k] * W[n,k] + bias[n]
// Block = 8 waves = 128(M) x 64(N) tile; each wave = 32x32 via 2x2 frags.
// mode 0: f32 [M,N]          (final output)
// mode 1: f16 head-major     qh[h][b][s][d]
// mode 2: f16 head-major-T   vt[h][b][d][s]
// ---------------------------------------------------------------------------
__device__ __forceinline__ void gemm_store(v8f acc, int mbase, int nbase,
                                           const float* __restrict__ bias,
                                           float* __restrict__ outF,
                                           _Float16* __restrict__ outH,
                                           int N, int mode, int l16, int half) {
  int col = nbase + l16;
  float bv = bias[col];
#pragma unroll
  for (int r = 0; r < 8; ++r) {
    int row = mbase + r + 8 * half;
    float v = acc[r] + bv;
    if (mode == 0) {
      outF[(size_t)row * N + col] = v;
    } else {
      int b = row >> 11, s = row & (SEQ - 1);
      int h = col >> 6, d = col & 63;
      if (mode == 1)
        outH[(((size_t)(h * BATCH + b)) * SEQ + s) * HEAD_DIM + d] = (_Float16)v;
      else
        outH[(((size_t)(h * BATCH + b)) * HEAD_DIM + d) * SEQ + s] = (_Float16)v;
    }
  }
}

__global__ __launch_bounds__(256) void gemm_wmma(const _Float16* __restrict__ A,
                                                 const _Float16* __restrict__ W,
                                                 const float* __restrict__ bias,
                                                 float* __restrict__ outF,
                                                 _Float16* __restrict__ outH,
                                                 int M, int N, int K, int mode) {
  const int lane = threadIdx.x & 31;
  const int wave = threadIdx.x >> 5;
  const int half = lane >> 4, l16 = lane & 15;
  const int m0 = blockIdx.y * 128 + (wave >> 1) * 32;
  const int n0 = blockIdx.x * 64 + (wave & 1) * 32;

  const _Float16* arow0 = A + (size_t)(m0 + l16) * K;
  const _Float16* arow1 = A + (size_t)(m0 + 16 + l16) * K;
  const _Float16* wrow0 = W + (size_t)(n0 + l16) * K;
  const _Float16* wrow1 = W + (size_t)(n0 + 16 + l16) * K;
  const int ho = half * 8;

  v8f c00 = {}, c01 = {}, c10 = {}, c11 = {};
  for (int k0 = 0; k0 < K; k0 += 32) {
    v16h a0 = frag_ld(arow0 + k0 + ho, arow0 + k0 + 16 + ho);
    v16h a1 = frag_ld(arow1 + k0 + ho, arow1 + k0 + 16 + ho);
    v16h b0 = frag_ld(wrow0 + k0 + ho, wrow0 + k0 + 16 + ho);
    v16h b1 = frag_ld(wrow1 + k0 + ho, wrow1 + k0 + 16 + ho);
    c00 = wmma16(a0, b0, c00);
    c01 = wmma16(a0, b1, c01);
    c10 = wmma16(a1, b0, c10);
    c11 = wmma16(a1, b1, c11);
  }
  gemm_store(c00, m0,      n0,      bias, outF, outH, N, mode, l16, half);
  gemm_store(c01, m0,      n0 + 16, bias, outF, outH, N, mode, l16, half);
  gemm_store(c10, m0 + 16, n0,      bias, outF, outH, N, mode, l16, half);
  gemm_store(c11, m0 + 16, n0 + 16, bias, outF, outH, N, mode, l16, half);
}

// ---------------------------------------------------------------------------
// Flash-style spike attention: one wave per 16-row tile of one (h,b).
// Streams keys 32/step: 4 WMMA scores + gate/mask/clip + online softmax
// + P through LDS (C-layout -> A-layout) + 4 WMMA for P@V.
// ---------------------------------------------------------------------------
__global__ __launch_bounds__(32) void attn_wmma(const _Float16* __restrict__ qh,
                                                const _Float16* __restrict__ kh,
                                                const _Float16* __restrict__ vt,
                                                const float* __restrict__ thr,
                                                const int* __restrict__ mask,
                                                _Float16* __restrict__ attn) {
  __shared__ _Float16 ldsP[16 * 32];
  const int lane = threadIdx.x & 31;
  const int half = lane >> 4, l16 = lane & 15;
  const int t = blockIdx.x;
  const int mt = t & 127;          // S/16 tiles
  const int b = (t >> 7) & (BATCH - 1);
  const int h = t >> 8;

  const size_t hb = (size_t)(h * BATCH + b);
  const _Float16* Q = qh + hb * SEQ * HEAD_DIM;
  const _Float16* Km = kh + hb * SEQ * HEAD_DIM;
  const _Float16* V = vt + hb * (size_t)HEAD_DIM * SEQ;
  const float* thrb = thr + b * SEQ;
  const int* maskb = mask + b * SEQ;

  const int m0 = mt * 16;
  const int ho = half * 8;

  const _Float16* qrow = Q + (size_t)(m0 + l16) * HEAD_DIM;
  v16h aQ0 = frag_ld(qrow + ho,      qrow + 16 + ho);   // hd 0..31
  v16h aQ1 = frag_ld(qrow + 32 + ho, qrow + 48 + ho);   // hd 32..63

  float maskm[8], mrow[8], lrow[8];
#pragma unroll
  for (int r = 0; r < 8; ++r) {
    maskm[r] = (float)maskb[m0 + r + 8 * half];
    mrow[r] = -1e30f;
    lrow[r] = 0.f;
  }
  v8f o[4] = {v8f{}, v8f{}, v8f{}, v8f{}};

  for (int kb = 0; kb < SEQ; kb += 32) {
    const _Float16* krow0 = Km + (size_t)(kb + l16) * HEAD_DIM;
    const _Float16* krow1 = Km + (size_t)(kb + 16 + l16) * HEAD_DIM;
    v16h bK00 = frag_ld(krow0 + ho,      krow0 + 16 + ho);
    v16h bK01 = frag_ld(krow0 + 32 + ho, krow0 + 48 + ho);
    v16h bK10 = frag_ld(krow1 + ho,      krow1 + 16 + ho);
    v16h bK11 = frag_ld(krow1 + 32 + ho, krow1 + 48 + ho);
    v8f s0 = {}, s1 = {};
    s0 = wmma16(aQ0, bK00, s0);
    s0 = wmma16(aQ1, bK01, s0);
    s1 = wmma16(aQ0, bK10, s1);
    s1 = wmma16(aQ1, bK11, s1);

    const int n0c = kb + l16, n1c = n0c + 16;
    const float th0 = thrb[n0c], th1 = thrb[n1c];
    const float mk0 = (float)maskb[n0c], mk1 = (float)maskb[n1c];

    float x0[8], x1[8];
#pragma unroll
    for (int r = 0; r < 8; ++r) {
      float sc = fminf(6.f, fmaxf(-6.f, s0[r] * 0.25f));
      float g = 1.f / (1.f + expf(-5.f * (sc - th0)));
      float md = sc * (1.f + 2.f * g);
      if (maskm[r] * mk0 == 0.f) md = -10000.f;
      x0[r] = fminf(30.f, fmaxf(-30.f, md));

      sc = fminf(6.f, fmaxf(-6.f, s1[r] * 0.25f));
      g = 1.f / (1.f + expf(-5.f * (sc - th1)));
      md = sc * (1.f + 2.f * g);
      if (maskm[r] * mk1 == 0.f) md = -10000.f;
      x1[r] = fminf(30.f, fmaxf(-30.f, md));
    }

    // online softmax; row = r + 8*half lives across the 16-lane half-wave
#pragma unroll
    for (int r = 0; r < 8; ++r) {
      float mx = fmaxf(x0[r], x1[r]);
      for (int off = 8; off; off >>= 1) mx = fmaxf(mx, __shfl_xor(mx, off, 16));
      float nm = fmaxf(mrow[r], mx);
      float alpha = expf(mrow[r] - nm);
      float p0 = expf(x0[r] - nm), p1 = expf(x1[r] - nm);
      float ps = p0 + p1;
      for (int off = 8; off; off >>= 1) ps += __shfl_xor(ps, off, 16);
      lrow[r] = lrow[r] * alpha + ps;
      mrow[r] = nm;
      o[0][r] *= alpha; o[1][r] *= alpha; o[2][r] *= alpha; o[3][r] *= alpha;
      ldsP[(r + 8 * half) * 32 + l16]      = (_Float16)p0;
      ldsP[(r + 8 * half) * 32 + 16 + l16] = (_Float16)p1;
    }
    asm volatile("s_wait_dscnt 0" ::: "memory");  // P stores visible before A-frag reads

    v16h aP = frag_ld(&ldsP[l16 * 32 + ho], &ldsP[l16 * 32 + 16 + ho]);

#pragma unroll
    for (int c = 0; c < 4; ++c) {
      const _Float16* vrow = V + (size_t)(c * 16 + l16) * SEQ + kb;
      v16h bV = frag_ld(vrow + ho, vrow + 16 + ho);
      o[c] = wmma16(aP, bV, o[c]);
    }
  }

#pragma unroll
  for (int r = 0; r < 8; ++r) {
    int row = m0 + r + 8 * half;
    float inv = 1.f / lrow[r];
    size_t base = ((size_t)b * SEQ + row) * D_MODEL + h * HEAD_DIM + l16;
    attn[base]      = (_Float16)(o[0][r] * inv);
    attn[base + 16] = (_Float16)(o[1][r] * inv);
    attn[base + 32] = (_Float16)(o[2][r] * inv);
    attn[base + 48] = (_Float16)(o[3][r] * inv);
  }
}

// ---------------------------------------------------------------------------
extern "C" void kernel_launch(void* const* d_in, const int* in_sizes, int n_in,
                              void* d_out, int out_size, void* d_ws, size_t ws_size,
                              hipStream_t stream) {
  (void)in_sizes; (void)n_in; (void)out_size; (void)ws_size;
  const float* query  = (const float*)d_in[0];
  const float* key_in = (const float*)d_in[1];
  const float* value  = (const float*)d_in[2];
  const int*   mask   = (const int*)d_in[3];
  const float* dfeat  = (const float*)d_in[4];
  const float* wq_r = (const float*)d_in[5];
  const float* wq_i = (const float*)d_in[6];
  const float* wq_j = (const float*)d_in[7];
  const float* wq_k = (const float*)d_in[8];
  const float* bq   = (const float*)d_in[9];
  const float* wk_r = (const float*)d_in[10];
  const float* wk_i = (const float*)d_in[11];
  const float* wk_j = (const float*)d_in[12];
  const float* wk_k = (const float*)d_in[13];
  const float* bk   = (const float*)d_in[14];
  const float* wv   = (const float*)d_in[15];
  const float* bv   = (const float*)d_in[16];
  const float* wo   = (const float*)d_in[17];
  const float* bo   = (const float*)d_in[18];
  const float* threshold = (const float*)d_in[19];
  const float* dm1_w = (const float*)d_in[20];
  const float* dm1_b = (const float*)d_in[21];
  const float* dm2_w = (const float*)d_in[22];
  const float* dm2_b = (const float*)d_in[23];

  size_t off = 0;
  auto alloc = [&](size_t bytes) -> void* {
    void* p = (char*)d_ws + off;
    off += (bytes + 255) & ~(size_t)255;
    return p;
  };
  const size_t WB = (size_t)D_MODEL * D_MODEL * sizeof(_Float16);   // 2 MB
  const size_t AB = (size_t)MTOT * D_MODEL * sizeof(_Float16);     // 8 MB
  _Float16* Wq16 = (_Float16*)alloc(WB);
  _Float16* Wk16 = (_Float16*)alloc(WB);
  _Float16* Wv16 = (_Float16*)alloc(WB);
  _Float16* Wo16 = (_Float16*)alloc(WB);
  _Float16* q16  = (_Float16*)alloc(AB);
  _Float16* k16  = (_Float16*)alloc(AB);
  _Float16* v16  = (_Float16*)alloc(AB);
  _Float16* qh   = (_Float16*)alloc(AB);
  _Float16* kh   = (_Float16*)alloc(AB);
  _Float16* vt   = (_Float16*)alloc(AB);
  float*    thr  = (float*)alloc((size_t)MTOT * sizeof(float));
  _Float16* attn = q16;  // reuse: q16 dead after the q projection GEMM

  const int NW = D_MODEL * D_MODEL;  // 1M weight elements
  const int NA = MTOT * D_MODEL;     // 4M activation elements

  build_qweight<<<(NW + 255) / 256, 256, 0, stream>>>(wq_r, wq_i, wq_j, wq_k, Wq16);
  build_qweight<<<(NW + 255) / 256, 256, 0, stream>>>(wk_r, wk_i, wk_j, wk_k, Wk16);
  cvt_h<<<(NW + 255) / 256, 256, 0, stream>>>(wv, Wv16, NW);
  cvt_h<<<(NW + 255) / 256, 256, 0, stream>>>(wo, Wo16, NW);
  cvt_h<<<(NA + 255) / 256, 256, 0, stream>>>(query,  q16, NA);
  cvt_h<<<(NA + 255) / 256, 256, 0, stream>>>(key_in, k16, NA);
  cvt_h<<<(NA + 255) / 256, 256, 0, stream>>>(value,  v16, NA);
  density_thr<<<(MTOT + 255) / 256, 256, 0, stream>>>(dfeat, dm1_w, dm1_b, dm2_w, dm2_b,
                                                      threshold, thr, MTOT);

  dim3 gg(D_MODEL / 64, MTOT / 128);  // (16, 32)
  gemm_wmma<<<gg, 256, 0, stream>>>(q16, Wq16, bq, nullptr, qh, MTOT, D_MODEL, D_MODEL, 1);
  gemm_wmma<<<gg, 256, 0, stream>>>(k16, Wk16, bk, nullptr, kh, MTOT, D_MODEL, D_MODEL, 1);
  gemm_wmma<<<gg, 256, 0, stream>>>(v16, Wv16, bv, nullptr, vt, MTOT, D_MODEL, D_MODEL, 2);

  attn_wmma<<<N_HEADS * BATCH * (SEQ / 16), 32, 0, stream>>>(qh, kh, vt, thr, mask, attn);

  gemm_wmma<<<gg, 256, 0, stream>>>(attn, Wo16, bo, (float*)d_out, nullptr,
                                    MTOT, D_MODEL, D_MODEL, 0);
}
